// GroupedQueryAttention_19026705122064
// MI455X (gfx1250) — matrix operand order repaired and checked by reading, not compiled, not run on hardware
//
#include <hip/hip_runtime.h>

// ---------------------------------------------------------------------------
// GQA forward for MI455X (gfx1250, wave32, WMMA).
// - all GEMMs + attention einsums via v_wmma_f32_16x16x32_bf16
// - tile staging via global_load_async_to_lds_b128 (ASYNCcnt) where layout
//   allows; transposed staging otherwise so every fragment read is b128
// ---------------------------------------------------------------------------

typedef __bf16 bf16;
typedef __attribute__((ext_vector_type(16))) __bf16 v16bf;
typedef __attribute__((ext_vector_type(8)))  float  v8f;

constexpr int D_MODEL = 4096;
constexpr int NQ    = 32;
constexpr int NKV   = 8;
constexpr int HD    = 128;
constexpr int BATCH = 2;
constexpr int SEQ   = 2048;
constexpr int MROWS = BATCH * SEQ;      // 4096 rows for all projections
constexpr int KVDIM = NKV * HD;         // 1024

// --- WMMA fragment k order used by this file (contiguous, wave32) -------------------
// A (16x32, 16-bit): lane group g=lane>>4, row M=lane&15;
//   element e -> K = e + 16g (this file's order, not the ISA table's)
// B (32x16, 16-bit): lane holds column N=lane&15; element e -> K = e + 16g
__device__ __forceinline__ int a_frag_k(int e, int g) {
  return e + 16 * g;
}

__device__ __forceinline__ v8f wmma_bf16x32(v16bf a, v16bf b, v8f c) {
  return __builtin_amdgcn_wmma_f32_16x16x32_bf16(
      false, a, false, b, (short)0, c, false, false);
}

// --- Async global -> LDS staging (GLOBAL_LOAD_ASYNC_TO_LDS_B128, ASYNCcnt) -
// Generic pointers to LDS carry the wave-relative LDS byte offset in their
// low 32 bits (aperture in the high bits), which is exactly what the VDST
// VGPR of the async op must hold.
__device__ __forceinline__ uint32_t lds_off(const void* p) {
  return (uint32_t)(uintptr_t)p;
}
__device__ __forceinline__ void async_load_b128(uint32_t ldsoff, uint32_t voff,
                                                uint64_t sbase) {
  asm volatile("global_load_async_to_lds_b128 %0, %1, %2"
               :: "v"(ldsoff), "v"(voff), "s"(sbase) : "memory");
}
__device__ __forceinline__ void wait_asynccnt0() {
  asm volatile("s_wait_asynccnt 0x0" ::: "memory");
}

// ---------------------------------------------------------------------------
// fp32 -> bf16 conversion
// ---------------------------------------------------------------------------
__global__ void cvt_f32_bf16(const float* __restrict__ in,
                             bf16* __restrict__ out, int n) {
  for (int i = blockIdx.x * blockDim.x + threadIdx.x; i < n;
       i += gridDim.x * blockDim.x)
    out[i] = (bf16)in[i];
}

// ---------------------------------------------------------------------------
// GEMM: C[M,N] = A[M,K] * B[K,N], bf16 in, f32 accumulate.
// Block tile 128x128, 8 waves (4 M x 2 N), wave tile 32x64 (8 accumulators),
// K-step 32, double-buffered LDS. A staged async row-major; B staged
// transposed (K-major) so B fragments are contiguous b128 reads.
// ---------------------------------------------------------------------------
template <typename OutT>
__global__ __launch_bounds__(256) void gemm_bf16_wmma(
    const bf16* __restrict__ A, const bf16* __restrict__ Bw,
    OutT* __restrict__ C, int M, int N, int K) {
  __shared__ bf16 As[2][128][40];    // [buf][m][k]  row = 64B data + pad
  __shared__ bf16 BsT[2][128][40];   // [buf][n][k]  transposed tile

  const int tid  = threadIdx.x;
  const int lane = tid & 31, w = tid >> 5;
  const int g = lane >> 4, lm = lane & 15;
  const int wm = w & 3, wn = w >> 2;
  const int m0 = blockIdx.y * 128, n0 = blockIdx.x * 128;
  const int Nu = N >> 1;

  const uint64_t abase = (uint64_t)(uintptr_t)(A + (size_t)m0 * K);

  // stage A tile (128x32 bf16 = 512 x 16B chunks) asynchronously
  auto issueA = [&](int buf, int k0) {
    for (int id = tid; id < 512; id += 256) {
      int r = id >> 2, c = id & 3;
      async_load_b128(lds_off(&As[buf][r][0]) + c * 16,
                      (uint32_t)(r * K * 2 + k0 * 2 + c * 16), abase);
    }
  };
  // stage B tile transposed: dword global loads, two b16 LDS stores
  auto stageB = [&](int buf, int k0) {
    const uint32_t* Bg =
        reinterpret_cast<const uint32_t*>(Bw) + (size_t)k0 * Nu + (n0 >> 1);
    for (int id = tid; id < 32 * 64; id += 256) {
      int kk = id >> 6, n2 = id & 63;
      uint32_t d = Bg[(size_t)kk * Nu + n2];
      *(uint16_t*)&BsT[buf][2 * n2][kk]     = (uint16_t)(d & 0xffffu);
      *(uint16_t*)&BsT[buf][2 * n2 + 1][kk] = (uint16_t)(d >> 16);
    }
  };

  v8f acc[2][4] = {};
  const int nsteps = K >> 5;

  issueA(0, 0);
  stageB(0, 0);
  wait_asynccnt0();
  __syncthreads();

  for (int s = 0; s < nsteps; ++s) {
    const int cb = s & 1;
    if (s + 1 < nsteps) {
      issueA(cb ^ 1, (s + 1) * 32);
      stageB(cb ^ 1, (s + 1) * 32);
    }

    v16bf afr[2];
#pragma unroll
    for (int a = 0; a < 2; ++a)
#pragma unroll
      for (int e = 0; e < 16; ++e)
        afr[a][e] = As[cb][wm * 32 + a * 16 + lm][a_frag_k(e, g)];
#pragma unroll
    for (int t = 0; t < 4; ++t) {
      v16bf bf_;
#pragma unroll
      for (int e = 0; e < 16; ++e)
        bf_[e] = BsT[cb][wn * 64 + t * 16 + lm][e + 16 * g];
#pragma unroll
      for (int a = 0; a < 2; ++a)
        acc[a][t] = wmma_bf16x32(afr[a], bf_, acc[a][t]);
    }

    wait_asynccnt0();
    __syncthreads();
  }

  // C/D layout: lane -> N = lane&15; VGPR r -> M = r + 8*(lane>>4)
#pragma unroll
  for (int a = 0; a < 2; ++a)
#pragma unroll
    for (int t = 0; t < 4; ++t)
#pragma unroll
      for (int r = 0; r < 8; ++r) {
        int row = m0 + wm * 32 + a * 16 + r + 8 * g;
        int col = n0 + wn * 64 + t * 16 + lm;
        C[(size_t)row * N + col] = (OutT)acc[a][t][r];
      }
}

// ---------------------------------------------------------------------------
// RoPE (interleaved pairs), in-place on bf16 (B,S,H,HD)
// ---------------------------------------------------------------------------
__global__ void rope_kernel(bf16* __restrict__ t, const float* __restrict__ fc,
                            const float* __restrict__ fs, int H, int total) {
  int idx = blockIdx.x * blockDim.x + threadIdx.x;
  if (idx >= total) return;
  int i = idx & 63;
  int s = ((idx >> 6) / H) % SEQ;
  size_t base = (size_t)(idx >> 6) * HD;
  float a = (float)t[base + 2 * i];
  float b = (float)t[base + 2 * i + 1];
  float c = fc[s * 64 + i], sn = fs[s * 64 + i];
  t[base + 2 * i]     = (bf16)(a * c - b * sn);
  t[base + 2 * i + 1] = (bf16)(a * sn + b * c);
}

// ---------------------------------------------------------------------------
// Flash attention: block = 4 waves = 64 queries; grid (S/64, NQ, B).
// K tile async-staged row-major (QK^T B-fragments are row-contiguous);
// V tile staged transposed so PV B-fragments are contiguous b128 reads.
// Online softmax; P re-laid out through LDS for the PV A-fragments.
// ---------------------------------------------------------------------------
__global__ __launch_bounds__(128) void attn_fwd(
    const bf16* __restrict__ q, const bf16* __restrict__ k,
    const bf16* __restrict__ v, bf16* __restrict__ o) {
  __shared__ bf16 Ks[64][136];     // [key][dim]
  __shared__ bf16 VsT[128][72];    // [dim][key]  (transposed)
  __shared__ bf16 Ps[4][16][72];   // per-wave probabilities

  const int tid  = threadIdx.x;
  const int lane = tid & 31, w = tid >> 5;
  const int g = lane >> 4, lm = lane & 15;
  const int qt = blockIdx.x;
  const int h  = blockIdx.y;
  const int b  = blockIdx.z;
  const int hk = h >> 2;
  const int q0 = qt * 64;
  const float scale = 0.08838834764831845f;   // 1/sqrt(128)

  // Q fragments: 16 rows x 128 dims = 4 A-fragments
  v16bf qf[4];
  {
    const int row = q0 + w * 16 + lm;
    const size_t base = (((size_t)b * SEQ + row) * NQ + h) * HD;
#pragma unroll
    for (int dt = 0; dt < 4; ++dt)
#pragma unroll
      for (int e = 0; e < 16; ++e)
        qf[dt][e] = q[base + dt * 32 + a_frag_k(e, g)];
  }

  float m_i[8], l_i[8];
  v8f oacc[8] = {};
#pragma unroll
  for (int r = 0; r < 8; ++r) { m_i[r] = -1e30f; l_i[r] = 0.f; }

  for (int kt = 0; kt <= qt; ++kt) {
    const size_t kvelem = (((size_t)b * SEQ + (size_t)kt * 64) * NKV + hk) * HD;
    // K tile: async 16B chunks (64 keys x 256B = 1024 chunks)
    {
      const uint64_t kb64 = (uint64_t)(uintptr_t)(k + kvelem);
      for (int id = tid; id < 1024; id += 128) {
        int r = id >> 4, c = id & 15;
        async_load_b128(lds_off(&Ks[r][0]) + c * 16,
                        (uint32_t)(r * KVDIM * 2 + c * 16), kb64);
      }
    }
    // V tile transposed: dword loads + two b16 stores
    {
      const uint32_t* vg = reinterpret_cast<const uint32_t*>(v) + (kvelem >> 1);
      for (int id = tid; id < 64 * 64; id += 128) {
        int r = id >> 6, n2 = id & 63;
        uint32_t d = vg[(size_t)r * (KVDIM >> 1) + n2];
        *(uint16_t*)&VsT[2 * n2][r]     = (uint16_t)(d & 0xffffu);
        *(uint16_t*)&VsT[2 * n2 + 1][r] = (uint16_t)(d >> 16);
      }
    }
    wait_asynccnt0();
    __syncthreads();

    // ---- S = Q K^T : 4 key sub-tiles x 4 d-chunks = 16 WMMAs ----
    v8f s[4] = {};
#pragma unroll
    for (int t = 0; t < 4; ++t)
#pragma unroll
      for (int dt = 0; dt < 4; ++dt) {
        v16bf bf_;
#pragma unroll
        for (int e = 0; e < 16; ++e)
          bf_[e] = Ks[t * 16 + lm][dt * 32 + e + 16 * g];
        s[t] = wmma_bf16x32(qf[dt], bf_, s[t]);
      }

    // ---- online softmax ----
    float p[4][8], rowm[8];
#pragma unroll
    for (int r = 0; r < 8; ++r) rowm[r] = -1e30f;
#pragma unroll
    for (int t = 0; t < 4; ++t) {
      const int key = kt * 64 + t * 16 + lm;
#pragma unroll
      for (int r = 0; r < 8; ++r) {
        const int row = q0 + w * 16 + r + 8 * g;
        float sv = s[t][r] * scale + (key > row ? -1e9f : 0.f);
        p[t][r] = sv;
        rowm[r] = fmaxf(rowm[r], sv);
      }
    }
#pragma unroll
    for (int r = 0; r < 8; ++r)
#pragma unroll
      for (int off = 1; off < 16; off <<= 1)
        rowm[r] = fmaxf(rowm[r], __shfl_xor(rowm[r], off, 32));

#pragma unroll
    for (int r = 0; r < 8; ++r) {
      const float newm = fmaxf(m_i[r], rowm[r]);
      const float corr = __expf(m_i[r] - newm);
      m_i[r] = newm;
      float rs = 0.f;
#pragma unroll
      for (int t = 0; t < 4; ++t) {
        p[t][r] = __expf(p[t][r] - newm);
        rs += p[t][r];
      }
#pragma unroll
      for (int off = 1; off < 16; off <<= 1)
        rs += __shfl_xor(rs, off, 32);
      l_i[r] = l_i[r] * corr + rs;
#pragma unroll
      for (int dt = 0; dt < 8; ++dt) oacc[dt][r] *= corr;
    }

    // ---- P -> LDS (row-major 16x64 per wave) ----
#pragma unroll
    for (int t = 0; t < 4; ++t)
#pragma unroll
      for (int r = 0; r < 8; ++r)
        Ps[w][r + 8 * g][t * 16 + lm] = (bf16)p[t][r];
    __syncthreads();

    // ---- O += P V : 2 key-chunks x 8 d-tiles = 16 WMMAs ----
#pragma unroll
    for (int kk = 0; kk < 2; ++kk) {
      v16bf af;
#pragma unroll
      for (int e = 0; e < 16; ++e)
        af[e] = Ps[w][lm][kk * 32 + a_frag_k(e, g)];
#pragma unroll
      for (int dt = 0; dt < 8; ++dt) {
        v16bf bf_;
#pragma unroll
        for (int e = 0; e < 16; ++e)
          bf_[e] = VsT[dt * 16 + lm][kk * 32 + e + 16 * g];
        oacc[dt] = wmma_bf16x32(af, bf_, oacc[dt]);
      }
    }
    __syncthreads();
  }

  // ---- normalize + store (B,S,NQ,HD) bf16 ----
#pragma unroll
  for (int r = 0; r < 8; ++r) {
    const float inv = 1.0f / l_i[r];
    const int row = q0 + w * 16 + r + 8 * g;
    const size_t base = (((size_t)b * SEQ + row) * NQ + h) * HD;
#pragma unroll
    for (int dt = 0; dt < 8; ++dt)
      o[base + dt * 16 + lm] = (bf16)(oacc[dt][r] * inv);
  }
}

// ---------------------------------------------------------------------------
// Host-side orchestration (graph-capture safe)
// ---------------------------------------------------------------------------
extern "C" void kernel_launch(void* const* d_in, const int* in_sizes, int n_in,
                              void* d_out, int out_size, void* d_ws,
                              size_t ws_size, hipStream_t stream) {
  (void)in_sizes; (void)n_in; (void)out_size; (void)ws_size;

  const float* x  = (const float*)d_in[0];
  const float* wq = (const float*)d_in[1];
  const float* wk = (const float*)d_in[2];
  const float* wv = (const float*)d_in[3];
  const float* wo = (const float*)d_in[4];
  const float* fc = (const float*)d_in[5];
  const float* fs = (const float*)d_in[6];
  // d_in[7]: mask — pure causal, applied analytically inside attn_fwd
  float* out = (float*)d_out;

  char* ws = (char*)d_ws;
  size_t off = 0;
  auto take = [&](size_t elems) {
    bf16* p = (bf16*)(ws + off);
    off += ((elems * sizeof(bf16)) + 255) & ~(size_t)255;
    return p;
  };
  bf16* xb  = take((size_t)MROWS * D_MODEL);
  bf16* wqb = take((size_t)D_MODEL * D_MODEL);
  bf16* wkb = take((size_t)D_MODEL * KVDIM);
  bf16* wvb = take((size_t)D_MODEL * KVDIM);
  bf16* wob = take((size_t)D_MODEL * D_MODEL);
  bf16* qb  = take((size_t)MROWS * D_MODEL);
  bf16* kb  = take((size_t)MROWS * KVDIM);
  bf16* vb  = take((size_t)MROWS * KVDIM);
  bf16* ob  = take((size_t)MROWS * D_MODEL);

  auto cvt = [&](const float* src, bf16* dst, size_t n) {
    cvt_f32_bf16<<<dim3(2048), dim3(256), 0, stream>>>(src, dst, (int)n);
  };
  cvt(x,  xb,  (size_t)MROWS * D_MODEL);
  cvt(wq, wqb, (size_t)D_MODEL * D_MODEL);
  cvt(wk, wkb, (size_t)D_MODEL * KVDIM);
  cvt(wv, wvb, (size_t)D_MODEL * KVDIM);
  cvt(wo, wob, (size_t)D_MODEL * D_MODEL);

  gemm_bf16_wmma<bf16><<<dim3(D_MODEL / 128, MROWS / 128), 256, 0, stream>>>(
      xb, wqb, qb, MROWS, D_MODEL, D_MODEL);
  gemm_bf16_wmma<bf16><<<dim3(KVDIM / 128, MROWS / 128), 256, 0, stream>>>(
      xb, wkb, kb, MROWS, KVDIM, D_MODEL);
  gemm_bf16_wmma<bf16><<<dim3(KVDIM / 128, MROWS / 128), 256, 0, stream>>>(
      xb, wvb, vb, MROWS, KVDIM, D_MODEL);

  {
    int totq = BATCH * SEQ * NQ * 64;
    rope_kernel<<<(totq + 255) / 256, 256, 0, stream>>>(qb, fc, fs, NQ, totq);
    int totk = BATCH * SEQ * NKV * 64;
    rope_kernel<<<(totk + 255) / 256, 256, 0, stream>>>(kb, fc, fs, NKV, totk);
  }

  attn_fwd<<<dim3(SEQ / 64, NQ, BATCH), 128, 0, stream>>>(qb, kb, vb, ob);

  gemm_bf16_wmma<float><<<dim3(D_MODEL / 128, MROWS / 128), 256, 0, stream>>>(
      ob, wob, out, MROWS, D_MODEL, D_MODEL);
}
